// ModulatedConv_40621800685875
// MI455X (gfx1250) — compile-verified, matrix operand order checked by reading, not run on hardware
//
#include <hip/hip_runtime.h>
#include <hip/hip_bf16.h>

typedef __bf16 bf16_t;
typedef __attribute__((ext_vector_type(16))) __bf16 v16bf;
typedef __attribute__((ext_vector_type(8)))  float  v8f;

#define CIN_  32
#define COUT_ 32
#define HW_   65536
#define W_    256

// ---------------- kernel 0: style modulation + zero stats ----------------
__global__ void ModConv_prep(const float* __restrict__ style,
                             const float* __restrict__ modw,
                             const float* __restrict__ modb,
                             float* __restrict__ ws) {
    int gid = blockIdx.x * 256 + threadIdx.x;           // 0..511 = (b,cin)
    if (gid < 512) {
        int b = gid >> 5, c = gid & 31;
        const float* st = style + (long)b * 512;
        const float* mw = modw  + (long)c * 512;
        float acc = 0.f;
        for (int d = 0; d < 512; ++d) acc += st[d] * mw[d];
        ws[gid]        = acc * 0.044194173824159216f + modb[c];  // 1/sqrt(512)
        ws[512 + gid]  = 0.f;   // sum
        ws[1024 + gid] = 0.f;   // sumsq
    }
}

// ---------------- kernel 1: implicit-GEMM conv via bf16 WMMA (hi/lo split) ----------------
__global__ __launch_bounds__(256, 1)
void ModConv_wmma(const float* __restrict__ in,
                  const float* __restrict__ wgt,
                  const float* __restrict__ ws_s,
                  float* __restrict__ out,
                  float* __restrict__ ws_stats) {
    __shared__ bf16_t s_whi[9 * 32 * 32];   // [tap][cout][cin]
    __shared__ bf16_t s_wlo[9 * 32 * 32];
    __shared__ float  s_sum[32];
    __shared__ float  s_sq[32];

    const int tid     = threadIdx.x;
    const int b       = blockIdx.x >> 7;    // 128 row-pairs per sample
    const int rowpair = blockIdx.x & 127;

    if (tid < 32) { s_sum[tid] = 0.f; s_sq[tid] = 0.f; }

    // Build modulated weights: w = weight * (1/sqrt(288)) * s[b][cin], split hi/lo bf16
    for (int i = tid; i < 9 * 32 * 32; i += 256) {
        int t  = i >> 10;
        int r  = i & 1023;
        int co = r >> 5;
        int ci = r & 31;
        float w = wgt[(co * 32 + ci) * 9 + t] * 0.05892556509887896f * ws_s[b * 32 + ci];
        bf16_t hi = (bf16_t)w;
        bf16_t lo = (bf16_t)(w - (float)hi);
        int off = (t * 32 + co) * 32 + ci;
        s_whi[off] = hi;
        s_wlo[off] = lo;
    }
    __syncthreads();

    const int lane = tid & 31;
    const int wave = tid >> 5;
    const int nl   = lane & 15;     // column / M index within tile
    const int kh   = lane >> 4;     // K-half select
    const int y    = rowpair * 2 + (wave >> 2);
    const int xq   = (wave & 3) * 64;

    v8f acc[2][4];
#pragma unroll
    for (int mt = 0; mt < 2; ++mt)
#pragma unroll
        for (int j = 0; j < 4; ++j) acc[mt][j] = (v8f)(0.f);

    const long inbase = (long)b * CIN_ * HW_;

    for (int t = 0; t < 9; ++t) {
        const int dy = t / 3 - 1, dx = t % 3 - 1;
        const int yy = y + dy;
        const bool rv = (yy >= 0) && (yy < 256);

        // A fragments: 16(cout) x 32(cin), ISA 16-bit A layout
        v16bf ahi[2], alo[2];
#pragma unroll
        for (int mt = 0; mt < 2; ++mt) {
            const int co   = mt * 16 + nl;
            const int base = (t * 32 + co) * 32;
#pragma unroll
            for (int e = 0; e < 16; ++e) {
                const int ci = (e < 8) ? (kh * 8 + e) : (16 + kh * 8 + (e - 8));
                ahi[mt][e] = s_whi[base + ci];
                alo[mt][e] = s_wlo[base + ci];
            }
        }

#pragma unroll
        for (int j = 0; j < 4; ++j) {
            const int  x = xq + j * 16 + nl + dx;
            const bool v = rv && (x >= 0) && (x < 256);
            // B fragment: 32(cin) x 16(pixels), same per-lane K packing as A
            v16bf bhi, blo;
#pragma unroll
            for (int e = 0; e < 16; ++e) {
                const int ci = (e < 8) ? (kh * 8 + e) : (16 + kh * 8 + (e - 8));
                float f = v ? in[inbase + (long)ci * HW_ + (long)yy * W_ + x] : 0.f;
                bf16_t h = (bf16_t)f;
                bhi[e] = h;
                blo[e] = (bf16_t)(f - (float)h);
            }
#pragma unroll
            for (int mt = 0; mt < 2; ++mt) {
                acc[mt][j] = __builtin_amdgcn_wmma_f32_16x16x32_bf16(
                    false, ahi[mt], false, bhi, (short)0, acc[mt][j], false, false);
                acc[mt][j] = __builtin_amdgcn_wmma_f32_16x16x32_bf16(
                    false, ahi[mt], false, blo, (short)0, acc[mt][j], false, false);
                acc[mt][j] = __builtin_amdgcn_wmma_f32_16x16x32_bf16(
                    false, alo[mt], false, bhi, (short)0, acc[mt][j], false, false);
            }
        }
    }

    // Store conv result + per-(b,cout) sum / sumsq
    const long outb = (long)b * COUT_ * HW_;
#pragma unroll
    for (int mt = 0; mt < 2; ++mt) {
#pragma unroll
        for (int jj = 0; jj < 8; ++jj) {
            const int co = mt * 16 + jj + kh * 8;   // C/D layout: VGPR jj -> M = jj + 8*kh
            float p = 0.f, q = 0.f;
#pragma unroll
            for (int j = 0; j < 4; ++j) {
                float vv = acc[mt][j][jj];
                const int x = xq + j * 16 + nl;
                out[outb + (long)co * HW_ + (long)y * W_ + x] = vv;
                p += vv;
                q += vv * vv;
            }
#pragma unroll
            for (int m = 1; m < 16; m <<= 1) {      // butterfly within each 16-lane half
                p += __shfl_xor(p, m, 32);
                q += __shfl_xor(q, m, 32);
            }
            if (nl == 0) {
                atomicAdd(&s_sum[co], p);
                atomicAdd(&s_sq[co], q);
            }
        }
    }
    __syncthreads();
    if (tid < 32) {
        atomicAdd(&ws_stats[512  + b * 32 + tid], s_sum[tid]);
        atomicAdd(&ws_stats[1024 + b * 32 + tid], s_sq[tid]);
    }
}

// ---------------- kernel 2: demodulate (divide by unbiased spatial std) ----------------
__global__ void ModConv_demod(float* __restrict__ out, const float* __restrict__ ws) {
    const int plane = blockIdx.x;                    // 0..511 = (b,cout)
    const float s = ws[512 + plane];
    const float q = ws[1024 + plane];
    const float N = 65536.f;
    const float var = (q - s * s / N) / (N - 1.f);   // ddof=1
    const float inv = rsqrtf(var);
    float4* p = (float4*)(out + (long)plane * HW_);
    for (int i = threadIdx.x; i < HW_ / 4; i += blockDim.x) {
        float4 v = p[i];
        v.x *= inv; v.y *= inv; v.z *= inv; v.w *= inv;
        p[i] = v;
    }
}

extern "C" void kernel_launch(void* const* d_in, const int* in_sizes, int n_in,
                              void* d_out, int out_size, void* d_ws, size_t ws_size,
                              hipStream_t stream) {
    const float* input  = (const float*)d_in[0];   // (16,32,256,256)
    const float* style  = (const float*)d_in[1];   // (16,512)
    const float* weight = (const float*)d_in[2];   // (1,32,32,3,3)
    const float* modw   = (const float*)d_in[3];   // (32,512)
    const float* modb   = (const float*)d_in[4];   // (32,)
    float* out = (float*)d_out;
    float* ws  = (float*)d_ws;                     // [0:512) s, [512:1024) sum, [1024:1536) sumsq

    ModConv_prep<<<2, 256, 0, stream>>>(style, modw, modb, ws);
    ModConv_wmma<<<16 * 128, 256, 0, stream>>>(input, weight, ws, out, ws);
    ModConv_demod<<<512, 256, 0, stream>>>(out, ws);
}
